// CLA_87488483820324
// MI455X (gfx1250) — matrix operand order repaired and checked by reading, not compiled, not run on hardware
//
#include <hip/hip_runtime.h>
#include <hip/hip_bf16.h>

// ---------------------------------------------------------------------------
// Constants
// ---------------------------------------------------------------------------
#define EPSV 1e-5f
#define HH   64
#define WW   64
#define HWP  4096          // 64*64
#define PH   66            // padded height
#define PW   72            // padded row stride (66 used)
#define PCH  4752          // 66*72 per channel
#define NB   8             // batch

typedef __attribute__((ext_vector_type(16))) __bf16 v16bf;
typedef __attribute__((ext_vector_type(8)))  float  v8f;
typedef __attribute__((ext_vector_type(4)))  unsigned int v4u;
typedef __attribute__((ext_vector_type(8)))  int v8i;
typedef __attribute__((ext_vector_type(4)))  int v4i;

struct Pk32 { uint4 a, b; };   // 32 bytes -> v16bf

static __device__ inline unsigned short f2b(float f) {
    unsigned int u = __builtin_bit_cast(unsigned int, f);
    u += 0x7FFFu + ((u >> 16) & 1u);       // round-to-nearest-even
    return (unsigned short)(u >> 16);
}

static __device__ inline v16bf ldfrag(const unsigned short* base, int off0, int off1) {
    uint4 q0 = *reinterpret_cast<const uint4*>(base + off0);
    uint4 q1 = *reinterpret_cast<const uint4*>(base + off1);
    Pk32 pk{q0, q1};
    return __builtin_bit_cast(v16bf, pk);
}

static __device__ inline int refl(int i) { return i < 0 ? -i : (i > 63 ? 126 - i : i); }

// ---------------------------------------------------------------------------
// Weight re-layout: OIHW f32 -> [Cout][(kh*3+kw)*Cin + ci] bf16
// ---------------------------------------------------------------------------
__global__ void relayout_conv_w(const float* __restrict__ w, unsigned short* __restrict__ dst,
                                int Cout, int Cin) {
    long long tot = (long long)Cout * Cin * 9;
    long long i = (long long)blockIdx.x * blockDim.x + threadIdx.x;
    if (i >= tot) return;
    int kc = 9 * Cin;
    int co = (int)(i / kc);
    int r  = (int)(i - (long long)co * kc);
    int p  = r / Cin;
    int ci = r - p * Cin;
    dst[i] = f2b(w[((long long)(co * Cin + ci)) * 9 + p]);
}

// f32 -> bf16 elementwise
__global__ void cvt_bf16(const float* __restrict__ s, unsigned short* __restrict__ d, long long n) {
    for (long long i = (long long)blockIdx.x * blockDim.x + threadIdx.x; i < n;
         i += (long long)gridDim.x * blockDim.x)
        d[i] = f2b(s[i]);
}

// ---------------------------------------------------------------------------
// Reflect-pad + convert to bf16.  dst[b][c][hp*72+wp].
// ---------------------------------------------------------------------------
__global__ void pad_bf16(const float* __restrict__ s0, const float* __restrict__ s1,
                         int csplit, int C, unsigned short* __restrict__ dst) {
    int b = blockIdx.z, c = blockIdx.x, t = threadIdx.x;
    const float* src = (c < csplit)
        ? s0 + ((long long)b * csplit + c) * HWP
        : s1 + ((long long)b * (C - csplit) + (c - csplit)) * HWP;
    unsigned short* dp = dst + ((long long)b * C + c) * PCH;
    for (int i = t; i < PH * PH; i += 256) {
        int hp = i / PH, wp = i - hp * PH;
        dp[hp * PW + wp] = f2b(src[refl(hp - 1) * WW + refl(wp - 1)]);
    }
}

// GroupNorm(1) affine (+optional ReLU) fused with reflect-pad + bf16 convert.
__global__ void gn_pad_bf16(const float* __restrict__ act, const float* __restrict__ stats,
                            const float* __restrict__ g, const float* __restrict__ be,
                            int relu, int C, unsigned short* __restrict__ dst) {
    int b = blockIdx.z, c = blockIdx.x, t = threadIdx.x;
    float mean = stats[b * 2], rstd = stats[b * 2 + 1];
    float gg = g[c] * rstd;
    float bb = be[c] - mean * gg;
    const float* src = act + ((long long)b * C + c) * HWP;
    unsigned short* dp = dst + ((long long)b * C + c) * PCH;
    for (int i = t; i < PH * PH; i += 256) {
        int hp = i / PH, wp = i - hp * PH;
        float v = src[refl(hp - 1) * WW + refl(wp - 1)] * gg + bb;
        if (relu) v = fmaxf(v, 0.f);
        dp[hp * PW + wp] = f2b(v);
    }
}

// ---------------------------------------------------------------------------
// GroupNorm(1,C) stats: two-pass deterministic reduction
// ---------------------------------------------------------------------------
__global__ void gn_partial(const float* __restrict__ act, long long NE, float* __restrict__ parts) {
    __shared__ float r1[256], r2[256];
    int t = threadIdx.x;
    long long base = (long long)blockIdx.z * NE;
    float s = 0.f, s2 = 0.f;
    for (long long i = (long long)blockIdx.x * 256 + t; i < NE; i += 65536) {
        float v = act[base + i]; s += v; s2 += v * v;
    }
    r1[t] = s; r2[t] = s2; __syncthreads();
    for (int o = 128; o > 0; o >>= 1) {
        if (t < o) { r1[t] += r1[t + o]; r2[t] += r2[t + o]; }
        __syncthreads();
    }
    if (t == 0) {
        long long pi = ((long long)blockIdx.z * 256 + blockIdx.x) * 2;
        parts[pi] = r1[0]; parts[pi + 1] = r2[0];
    }
}

__global__ void gn_final(const float* __restrict__ parts, float* __restrict__ stats, float NE) {
    __shared__ float r1[256], r2[256];
    int b = blockIdx.x, t = threadIdx.x;
    r1[t] = parts[(b * 256 + t) * 2];
    r2[t] = parts[(b * 256 + t) * 2 + 1];
    __syncthreads();
    for (int o = 128; o > 0; o >>= 1) {
        if (t < o) { r1[t] += r1[t + o]; r2[t] += r2[t + o]; }
        __syncthreads();
    }
    if (t == 0) {
        float m = r1[0] / NE;
        float v = r2[0] / NE - m * m;
        stats[b * 2] = m;
        stats[b * 2 + 1] = rsqrtf(v + EPSV);
    }
}

// ---------------------------------------------------------------------------
// Implicit-GEMM 3x3 conv, bf16 WMMA 16x16x32, f32 accumulate.
// Double-buffered LDS, K-step 64.  A-tile streamed with the CDNA5 async path
// (global_load_async_to_lds_b128, ASYNCcnt) -> zero VGPR staging for A.
// B-tile: im2col pairs of K-rows packed to dwords (8x ds_store_b32/thread).
// grid = (64 rows, Cout/128, B), block 256 (8 waves)
// ---------------------------------------------------------------------------
__global__ __launch_bounds__(256) void conv3x3_wmma(
    const unsigned short* __restrict__ Wr, const unsigned short* __restrict__ pad,
    float* __restrict__ out, const float* __restrict__ bias, int Cout, int Cin) {
    __shared__ unsigned short lA[2][128 * 64];
    __shared__ unsigned short lB[2][64 * 64];
    const int K  = 9 * Cin;
    const int h  = blockIdx.x;
    const int m0 = blockIdx.y * 128;
    const int b  = blockIdx.z;
    const int t  = threadIdx.x;
    const int lane = t & 31, wv = t >> 5;
    const int wm = wv >> 2, wn = wv & 3;
    const int hi = (lane >> 4) & 1, lm = lane & 15;
    const unsigned short* padB = pad + (long long)b * Cin * PCH;

    const int arow = t >> 1, ahalf = (t & 1) * 32;   // A: 64B async per thread
    const int bkk  = (t >> 3) * 2, bnb = (t & 7) * 8; // B: 2 K-rows x 8 n per thread

    unsigned rbp[8];

    auto asyncA = [&](int k0, int bf) {
        const unsigned short* gp = Wr + (long long)(m0 + arow) * K + k0 + ahalf;
        unsigned lofs = (unsigned)(unsigned long long)(const void*)&lA[bf][arow * 64 + ahalf];
        #pragma unroll
        for (int i = 0; i < 4; i++) {
            asm volatile("global_load_async_to_lds_b128 %0, %1, off"
                         :: "v"(lofs + i * 16), "v"(gp + i * 8) : "memory");
        }
    };
    auto loadB = [&](int k0) {
        int p  = k0 / Cin;
        int ci = (k0 - p * Cin) + bkk;
        int kh = p / 3, kw = p - kh * 3;
        const unsigned short* s0 = padB + (long long)ci * PCH + (h + kh) * PW + kw + bnb;
        const unsigned short* s1 = s0 + PCH;
        #pragma unroll
        for (int j = 0; j < 8; j++)
            rbp[j] = (unsigned)s0[j] | ((unsigned)s1[j] << 16);
    };
    auto stB = [&](int bf) {
        #pragma unroll
        for (int j = 0; j < 8; j++)
            *reinterpret_cast<unsigned*>(&lB[bf][(bnb + j) * 64 + bkk]) = rbp[j];
    };

    v8f acc[4] = {};
    asyncA(0, 0); loadB(0); stB(0);
    asm volatile("s_wait_asynccnt 0" ::: "memory");
    int buf = 0;
    for (int k0 = 0; k0 < K; k0 += 64) {
        __syncthreads();                       // lds[buf] published
        const int kn = k0 + 64;
        if (kn < K) {
            asyncA(kn, buf ^ 1);               // async into the idle buffer
            loadB(kn);                         // overlap with WMMA below
        }
        #pragma unroll
        for (int s = 0; s < 2; s++) {
            const int kb = s * 32;
            int bro = (wn * 16 + lm) * 64 + kb + (hi ? 16 : 0);
            v16bf bfr = ldfrag(&lB[buf][0], bro, bro + 8);
            #pragma unroll
            for (int i = 0; i < 4; i++) {
                int row = (wm * 64 + i * 16 + lm) * 64 + kb;
                v16bf af = ldfrag(&lA[buf][0], row + (hi ? 16 : 0), row + (hi ? 24 : 8));
                acc[i] = __builtin_amdgcn_wmma_f32_16x16x32_bf16(
                    false, af, false, bfr, (short)0, acc[i], false, false);
            }
        }
        if (kn < K) {
            __syncthreads();                   // all readers of lds[buf^1] retired
            stB(buf ^ 1);
            asm volatile("s_wait_asynccnt 0" ::: "memory");
            buf ^= 1;
        }
    }
    float* outB = out + (long long)b * Cout * HWP;
    int nn = h * WW + wn * 16 + lm;
    #pragma unroll
    for (int i = 0; i < 4; i++) {
        int mb = m0 + wm * 64 + i * 16 + hi * 8;
        #pragma unroll
        for (int r = 0; r < 8; r++)
            outB[(long long)(mb + r) * HWP + nn] = acc[i][r] + bias[mb + r];
    }
}

// ---------------------------------------------------------------------------
// Generic bf16 WMMA GEMM: C[b] = A[b](MxK) * B[b](KxN) (+bias per-row)
// Double-buffered LDS, K-step 64.  A-tile moved by the Tensor Data Mover:
// one wave0-issued tensor_load_to_lds per K-step (128x64 bf16 2D tile,
// TENSORcnt-tracked); B staged via packed dword regs.
// grid = (N/64, M/128, batch)
// ---------------------------------------------------------------------------
__global__ __launch_bounds__(256) void gemm_bf16_wmma(
    const unsigned short* __restrict__ A, long long sA,
    const unsigned short* __restrict__ B, long long sB,
    float* __restrict__ C, long long sC,
    const float* __restrict__ bias, int N, int K) {
    __shared__ unsigned short lA[2][128 * 64];
    __shared__ unsigned short lB[2][64 * 64];
    const int n0 = blockIdx.x * 64;
    const int m0 = blockIdx.y * 128;
    const unsigned short* Ab = A + (long long)blockIdx.z * sA;
    const unsigned short* Bb = B + (long long)blockIdx.z * sB;
    float* Cb = C + (long long)blockIdx.z * sC;
    const int t = threadIdx.x;
    const int lane = t & 31, wv = t >> 5;
    const int wm = wv >> 2, wn = wv & 3;
    const int hi = (lane >> 4) & 1, lm = lane & 15;

    const int bkk = (t >> 3) * 2, bnb = (t & 7) * 8;

    unsigned rbp[8];

    // One TDM descriptor moves the whole 128x64 bf16 A tile into lds[bf].
    // D# per CDNA5 ISA (2D, data_size=2B).
    auto tdmA = [&](int k0, int bf) {
        unsigned long long ga =
            (unsigned long long)(const void*)(Ab + (long long)m0 * K + k0);
        unsigned lofs = (unsigned)(unsigned long long)(const void*)&lA[bf][0];
        v4u g0 = { 1u,                                   // count=1, user mode
                   lofs,                                 // lds_addr
                   (unsigned)(ga & 0xffffffffu),         // global_addr[31:0]
                   (unsigned)((ga >> 32) & 0x01ffffffu) | (2u << 30) }; // type=2
        v8i g1 = { (int)0x00010000u,                     // data_size=1 -> 2 bytes
                   (int)(((unsigned)K & 0xffffu) << 16), // tensor_dim0[15:0]
                   (int)(((unsigned)K >> 16) & 0xffffu), // tensor_dim0[31:16], dim1 lo=0
                   (int)(1u | (64u << 16)),              // dim1 hi (=65536), tile_dim0=64
                   (int)128u,                            // tile_dim1=128, tile_dim2=0
                   (int)(unsigned)K,                     // tensor_dim0_stride[31:0]
                   0, 0 };
        v4i gz4 = { 0, 0, 0, 0 };
        v8i gz8 = { 0, 0, 0, 0, 0, 0, 0, 0 };
        __builtin_amdgcn_tensor_load_to_lds(g0, g1, gz4, gz4, gz8, 0);
    };
    auto loadBg = [&](int k0) {
        const unsigned short* gp = Bb + (long long)(k0 + bkk) * N + n0 + bnb;
        uint4 r0 = *reinterpret_cast<const uint4*>(gp);
        uint4 r1 = *reinterpret_cast<const uint4*>(gp + N);
        unsigned short u0[8], u1[8];
        *reinterpret_cast<uint4*>(u0) = r0;
        *reinterpret_cast<uint4*>(u1) = r1;
        #pragma unroll
        for (int j = 0; j < 8; j++)
            rbp[j] = (unsigned)u0[j] | ((unsigned)u1[j] << 16);
    };
    auto stB = [&](int bf) {
        #pragma unroll
        for (int j = 0; j < 8; j++)
            *reinterpret_cast<unsigned*>(&lB[bf][(bnb + j) * 64 + bkk]) = rbp[j];
    };

    v8f acc[4] = {};
    if (wv == 0) tdmA(0, 0);
    loadBg(0); stB(0);
    if (wv == 0) __builtin_amdgcn_s_wait_tensorcnt(0);
    int buf = 0;
    for (int k0 = 0; k0 < K; k0 += 64) {
        __syncthreads();                       // lds[buf] published (incl. TDM tile)
        const int kn = k0 + 64;
        if (kn < K) {
            if (wv == 0) tdmA(kn, buf ^ 1);    // TDM into the idle buffer
            loadBg(kn);
            if (kn + 64 < K)
                __builtin_prefetch(Bb + (long long)(kn + 64 + bkk) * N + n0 + bnb, 0, 1);
        }
        #pragma unroll
        for (int s = 0; s < 2; s++) {
            const int kb = s * 32;
            int bro = (wn * 16 + lm) * 64 + kb + (hi ? 16 : 0);
            v16bf bfr = ldfrag(&lB[buf][0], bro, bro + 8);
            #pragma unroll
            for (int i = 0; i < 4; i++) {
                int row = (wm * 64 + i * 16 + lm) * 64 + kb;
                v16bf af = ldfrag(&lA[buf][0], row + (hi ? 16 : 0), row + (hi ? 24 : 8));
                acc[i] = __builtin_amdgcn_wmma_f32_16x16x32_bf16(
                    false, af, false, bfr, (short)0, acc[i], false, false);
            }
        }
        if (kn < K) {
            __syncthreads();                   // readers of lds[buf^1] retired
            stB(buf ^ 1);
            if (wv == 0) __builtin_amdgcn_s_wait_tensorcnt(0);
            buf ^= 1;
        }
    }
    int nn = n0 + wn * 16 + lm;
    #pragma unroll
    for (int i = 0; i < 4; i++) {
        int mb = m0 + wm * 64 + i * 16 + hi * 8;
        #pragma unroll
        for (int r = 0; r < 8; r++) {
            float v = acc[i][r];
            if (bias) v += bias[mb + r];
            Cb[(long long)(mb + r) * N + nn] = v;
        }
    }
}

// ---------------------------------------------------------------------------
// AdaptiveAvgPool(EMB=64) fused with GN2 affine: emb[b*C+c][e]
// ---------------------------------------------------------------------------
__global__ void pool_affine(const float* __restrict__ act, const float* __restrict__ stats,
                            const float* __restrict__ g, const float* __restrict__ be,
                            int C, float* __restrict__ emb) {
    int b = blockIdx.z, c = blockIdx.x, e = threadIdx.x;
    float mean = stats[b * 2], rstd = stats[b * 2 + 1];
    const float* src = act + ((long long)b * C + c) * HWP + e * 64;
    float s = 0.f;
    for (int j = 0; j < 64; j++) s += src[j];
    s *= (1.f / 64.f);
    emb[((long long)b * C + c) * 64 + e] = (s - mean) * rstd * g[c] + be[c];
}

// ---------------------------------------------------------------------------
// MLP: x = LN2(lin2(relu(LN1(lin1(x)))))  in-place on [rows][64]
// ---------------------------------------------------------------------------
__global__ void mlp64(float* __restrict__ x,
                      const float* __restrict__ W1, const float* __restrict__ b1,
                      const float* __restrict__ g1, const float* __restrict__ be1,
                      const float* __restrict__ W2, const float* __restrict__ b2,
                      const float* __restrict__ g2, const float* __restrict__ be2) {
    __shared__ float sh[64];
    int t = threadIdx.x;
    float* xr = x + (long long)blockIdx.x * 64;
    sh[t] = xr[t]; __syncthreads();
    float a = b1[t];
    for (int k = 0; k < 64; k++) a += sh[k] * W1[t * 64 + k];
    __syncthreads(); sh[t] = a; __syncthreads();
    float m = 0.f; for (int k = 0; k < 64; k++) m += sh[k]; m *= (1.f / 64.f);
    float v = 0.f; for (int k = 0; k < 64; k++) { float d = sh[k] - m; v += d * d; } v *= (1.f / 64.f);
    a = fmaxf((a - m) * rsqrtf(v + EPSV) * g1[t] + be1[t], 0.f);
    __syncthreads(); sh[t] = a; __syncthreads();
    float c = b2[t];
    for (int k = 0; k < 64; k++) c += sh[k] * W2[t * 64 + k];
    __syncthreads(); sh[t] = c; __syncthreads();
    m = 0.f; for (int k = 0; k < 64; k++) m += sh[k]; m *= (1.f / 64.f);
    v = 0.f; for (int k = 0; k < 64; k++) { float d = sh[k] - m; v += d * d; } v *= (1.f / 64.f);
    xr[t] = (c - m) * rsqrtf(v + EPSV) * g2[t] + be2[t];
}

// y[row] = x[row] @ W^T + b   (64x64)
__global__ void proj64(const float* __restrict__ x, const float* __restrict__ W,
                       const float* __restrict__ b, float* __restrict__ y) {
    __shared__ float sh[64];
    int t = threadIdx.x;
    long long row = blockIdx.x;
    sh[t] = x[row * 64 + t]; __syncthreads();
    float a = b[t];
    for (int k = 0; k < 64; k++) a += sh[k] * W[t * 64 + k];
    y[row * 64 + t] = a;
}

// h[b][j] = b1[j] + dot(catflat[b], W1[j]); grid=(256,1,8)
__global__ void wpath1(const float* __restrict__ catf, const float* __restrict__ W1,
                       const float* __restrict__ b1, float* __restrict__ h) {
    __shared__ float red[256];
    int t = threadIdx.x, j = blockIdx.x, b = blockIdx.z;
    const float* xr = catf + (long long)b * 32768;
    const float* wr = W1 + (long long)j * 32768;
    float s = 0.f;
    for (int i = t; i < 32768; i += 256) s += xr[i] * wr[i];
    red[t] = s; __syncthreads();
    for (int o = 128; o > 0; o >>= 1) { if (t < o) red[t] += red[t + o]; __syncthreads(); }
    if (t == 0) h[(long long)b * 256 + j] = red[0] + b1[j];
}

// out[b][o] = b2[o] + h[b] . W2[o];  o < 131072; grid=(512,1,8)
__global__ void wpath2(const float* __restrict__ h, const float* __restrict__ W2,
                       const float* __restrict__ b2, float* __restrict__ out) {
    __shared__ float shh[256];
    int t = threadIdx.x, b = blockIdx.z;
    shh[t] = h[(long long)b * 256 + t]; __syncthreads();
    long long o = (long long)blockIdx.x * 256 + t;
    const float* wr = W2 + o * 256;
    float a = b2[o];
    for (int k = 0; k < 256; k++) a += shh[k] * wr[k];
    out[(long long)b * 131072 + o] = a;
}

// catf_re bf16: [b][512][4096]; d<256 -> vif, else irf
__global__ void catbf_build(const float* __restrict__ vif, const float* __restrict__ irf,
                            unsigned short* __restrict__ dst) {
    const long long total = (long long)NB * 512 * HWP;
    for (long long i = (long long)blockIdx.x * blockDim.x + threadIdx.x; i < total;
         i += (long long)gridDim.x * blockDim.x) {
        long long b = i >> 21;           // 512*4096
        long long r = i & 2097151;
        int d = (int)(r >> 12), s = (int)(r & 4095);
        float v = (d < 256) ? vif[((b << 8) + d) * HWP + s]
                            : irf[((b << 8) + (d - 256)) * HWP + s];
        dst[i] = f2b(v);
    }
}

// scores + softmax + bf16 attn: grid=(256,1,8), block 256 (each thread 2 cols)
__global__ void attn_softmax(const float* __restrict__ q, const float* __restrict__ k,
                             const float* __restrict__ wmat, unsigned short* __restrict__ attn) {
    __shared__ float sq[64];
    __shared__ float red[256];
    int t = threadIdx.x, c = blockIdx.x, b = blockIdx.z;
    if (t < 64) sq[t] = q[((long long)b * 256 + c) * 64 + t];
    __syncthreads();
    const float scale = 0.125f;   // 1/sqrt(64)
    float s[2];
    #pragma unroll
    for (int hlf = 0; hlf < 2; hlf++) {
        int d = t + hlf * 256;
        const float* kr = k + ((long long)b * 512 + d) * 64;
        float a = 0.f;
        for (int e = 0; e < 64; e++) a += sq[e] * kr[e];
        s[hlf] = (a + wmat[((long long)b * 256 + c) * 512 + d]) * scale;
    }
    red[t] = fmaxf(s[0], s[1]); __syncthreads();
    for (int o = 128; o > 0; o >>= 1) { if (t < o) red[t] = fmaxf(red[t], red[t + o]); __syncthreads(); }
    float mx = red[0]; __syncthreads();
    float e0 = __expf(s[0] - mx), e1 = __expf(s[1] - mx);
    red[t] = e0 + e1; __syncthreads();
    for (int o = 128; o > 0; o >>= 1) { if (t < o) red[t] += red[t + o]; __syncthreads(); }
    float inv = 1.f / red[0];
    unsigned short* ar = attn + ((long long)b * 256 + c) * 512;
    ar[t]       = f2b(e0 * inv);
    ar[t + 256] = f2b(e1 * inv);
}

// InstanceNorm (no affine) in place: grid=(512,1,8), block 256, 16 elems/thread
__global__ void instnorm(float* __restrict__ y) {
    __shared__ float r1[256], r2[256];
    int t = threadIdx.x;
    long long base = ((long long)blockIdx.z * 512 + blockIdx.x) * HWP;
    float vals[16];
    float s = 0.f, s2 = 0.f;
    #pragma unroll
    for (int i = 0; i < 16; i++) {
        float v = y[base + i * 256 + t];
        vals[i] = v; s += v; s2 += v * v;
    }
    r1[t] = s; r2[t] = s2; __syncthreads();
    for (int o = 128; o > 0; o >>= 1) { if (t < o) { r1[t] += r1[t + o]; r2[t] += r2[t + o]; } __syncthreads(); }
    float m = r1[0] * (1.f / HWP);
    float v = r2[0] * (1.f / HWP) - m * m;
    float rs = rsqrtf(v + EPSV);
    #pragma unroll
    for (int i = 0; i < 16; i++) y[base + i * 256 + t] = (vals[i] - m) * rs;
}

// ---------------------------------------------------------------------------
// Host launcher
// ---------------------------------------------------------------------------
extern "C" void kernel_launch(void* const* d_in, const int* in_sizes, int n_in,
                              void* d_out, int out_size, void* d_ws, size_t ws_size,
                              hipStream_t stream) {
    (void)in_sizes; (void)n_in; (void)out_size; (void)ws_size;
    // Flat input order (nested dicts depth-first, insertion order):
    // 0 vif, 1 irf, 2..17 vi branch, 18..33 ir branch, 34..49 cat branch,
    // 50..57 q/k proj, 58..65 wvi/wir MLPs, 66 out_w, 67 out_b
    enum { VI = 2, IR = 18, CAT = 34,
           QVIW = 50, QVIB, QIRW, QIRB, KVIW, KVIB, KIRW, KIRB,
           WVI1W = 58, WVI1B, WVI2W, WVI2B, WIR1W, WIR1B, WIR2W, WIR2B,
           OUTW = 66, OUTB = 67 };
    const float* vif = (const float*)d_in[0];
    const float* irf = (const float*)d_in[1];
    auto F = [&](int i) { return (const float*)d_in[i]; };

    char* ws = (char*)d_ws;
    unsigned short* PAD  = (unsigned short*)(ws);                  // 40 MiB slot (bf16 padded acts / catf_re)
    float*          ACT0 = (float*)(ws + (40LL  << 20));           // 64 MiB (conv1 out / y f32)
    float*          ACT1 = (float*)(ws + (108LL << 20));           // 64 MiB (conv2 out / y bf16)
    unsigned short* WB   = (unsigned short*)(ws + (176LL << 20));  // 16 MiB bf16 weights
    float*          EMBB = (float*)(ws + (192LL << 20));           // embeddings/projections
    float*          WMAT = (float*)(ws + (198LL << 20));           // w_vi/w_ir bias matrices
    unsigned short* ATTN = (unsigned short*)(ws + (206LL << 20));  // attn bf16
    float*          PARTS = (float*)(ws + (210LL << 20));
    float*          STATS = PARTS + 8 * 256 * 2;

    unsigned short* wb_vi1 = WB;
    unsigned short* wb_vi2 = WB + 589824;
    unsigned short* wb_ir1 = WB + 1179648;
    unsigned short* wb_ir2 = WB + 1769472;
    unsigned short* wb_ca1 = WB + 2359296;
    unsigned short* wb_ca2 = WB + 4718592;
    unsigned short* wb_out = WB + 7077888;

    float* vi_emb = EMBB;
    float* ir_emb = EMBB + 131072;
    float* cat_emb = EMBB + 262144;
    float* q_vi = EMBB + 524288;
    float* q_ir = EMBB + 655360;
    float* k_vi = EMBB + 786432;
    float* k_ir = EMBB + 1048576;
    float* h_vi = EMBB + 1310720;
    float* h_ir = h_vi + 2048;
    float* w_vi = WMAT;
    float* w_ir = WMAT + 1048576;
    unsigned short* attn_vi = ATTN;
    unsigned short* attn_ir = ATTN + 1048576;
    unsigned short* CATBF = PAD;                   // alias: pads dead after convs
    unsigned short* YBF   = (unsigned short*)ACT1; // alias: conv2 out dead after pooling

    // ---- weight prep (bf16) ----
    auto relay = [&](int idx, unsigned short* dst, int C) {
        long long tot = (long long)C * C * 9;
        relayout_conv_w<<<dim3((unsigned)((tot + 255) / 256)), 256, 0, stream>>>(F(idx), dst, C, C);
    };
    relay(VI + 0, wb_vi1, 256);  relay(VI + 4, wb_vi2, 256);
    relay(IR + 0, wb_ir1, 256);  relay(IR + 4, wb_ir2, 256);
    relay(CAT + 0, wb_ca1, 512); relay(CAT + 4, wb_ca2, 512);
    cvt_bf16<<<1024, 256, 0, stream>>>(F(OUTW), wb_out, 262144);

    // ---- branch pipeline ----
    auto branch = [&](const float* s0, const float* s1, int csplit, int C, int pb,
                      unsigned short* w1, unsigned short* w2, float* emb) {
        long long NE = (long long)C * HWP;
        pad_bf16<<<dim3(C, 1, NB), 256, 0, stream>>>(s0, s1, csplit, C, PAD);
        conv3x3_wmma<<<dim3(64, C / 128, NB), 256, 0, stream>>>(w1, PAD, ACT0, F(pb + 1), C, C);
        gn_partial<<<dim3(256, 1, NB), 256, 0, stream>>>(ACT0, NE, PARTS);
        gn_final<<<NB, 256, 0, stream>>>(PARTS, STATS, (float)NE);
        gn_pad_bf16<<<dim3(C, 1, NB), 256, 0, stream>>>(ACT0, STATS, F(pb + 2), F(pb + 3), 1, C, PAD);
        conv3x3_wmma<<<dim3(64, C / 128, NB), 256, 0, stream>>>(w2, PAD, ACT1, F(pb + 5), C, C);
        gn_partial<<<dim3(256, 1, NB), 256, 0, stream>>>(ACT1, NE, PARTS);
        gn_final<<<NB, 256, 0, stream>>>(PARTS, STATS, (float)NE);
        pool_affine<<<dim3(C, 1, NB), 64, 0, stream>>>(ACT1, STATS, F(pb + 6), F(pb + 7), C, emb);
        mlp64<<<NB * C, 64, 0, stream>>>(emb, F(pb + 8), F(pb + 9), F(pb + 10), F(pb + 11),
                                         F(pb + 12), F(pb + 13), F(pb + 14), F(pb + 15));
    };
    branch(vif, vif, 256, 256, VI, wb_vi1, wb_vi2, vi_emb);
    branch(irf, irf, 256, 256, IR, wb_ir1, wb_ir2, ir_emb);
    branch(vif, irf, 256, 512, CAT, wb_ca1, wb_ca2, cat_emb);

    // ---- projections ----
    proj64<<<NB * 256, 64, 0, stream>>>(vi_emb, F(QVIW), F(QVIB), q_vi);
    proj64<<<NB * 256, 64, 0, stream>>>(ir_emb, F(QIRW), F(QIRB), q_ir);
    proj64<<<NB * 512, 64, 0, stream>>>(cat_emb, F(KVIW), F(KVIB), k_vi);
    proj64<<<NB * 512, 64, 0, stream>>>(cat_emb, F(KIRW), F(KIRB), k_ir);

    // ---- dynamic bias matrices ----
    wpath1<<<dim3(256, 1, NB), 256, 0, stream>>>(cat_emb, F(WVI1W), F(WVI1B), h_vi);
    wpath2<<<dim3(512, 1, NB), 256, 0, stream>>>(h_vi, F(WVI2W), F(WVI2B), w_vi);
    wpath1<<<dim3(256, 1, NB), 256, 0, stream>>>(cat_emb, F(WIR1W), F(WIR1B), h_ir);
    wpath2<<<dim3(512, 1, NB), 256, 0, stream>>>(h_ir, F(WIR2W), F(WIR2B), w_ir);

    // ---- attention ----
    catbf_build<<<8192, 256, 0, stream>>>(vif, irf, CATBF);
    attn_softmax<<<dim3(256, 1, NB), 256, 0, stream>>>(q_vi, k_ir, w_vi, attn_vi);
    attn_softmax<<<dim3(256, 1, NB), 256, 0, stream>>>(q_ir, k_vi, w_ir, attn_ir);

    // feat = attn @ catf_re -> y (f32) in ACT0; channels [0:256) vi, [256:512) ir
    gemm_bf16_wmma<<<dim3(64, 2, NB), 256, 0, stream>>>(
        attn_vi, 256LL * 512, CATBF, 512LL * HWP, ACT0, 512LL * HWP, nullptr, HWP, 512);
    gemm_bf16_wmma<<<dim3(64, 2, NB), 256, 0, stream>>>(
        attn_ir, 256LL * 512, CATBF, 512LL * HWP, ACT0 + 256LL * HWP, 512LL * HWP, nullptr, HWP, 512);

    // ---- final 1x1 conv + InstanceNorm ----
    cvt_bf16<<<8192, 256, 0, stream>>>(ACT0, YBF, (long long)NB * 512 * HWP);
    gemm_bf16_wmma<<<dim3(64, 4, NB), 256, 0, stream>>>(
        wb_out, 0, YBF, 512LL * HWP, (float*)d_out, 512LL * HWP, F(OUTB), HWP, 512);
    instnorm<<<dim3(512, 1, NB), 256, 0, stream>>>((float*)d_out);
}
